// MoELayer_23854248362852
// MI455X (gfx1250) — matrix operand order repaired and checked by reading, not compiled, not run on hardware
//
#include <hip/hip_runtime.h>

// ---------------- problem constants ----------------
constexpr int E      = 8;      // routed experts
constexpr int H      = 2048;
constexpr int I_DIM  = 5120;
constexpr int NSH    = 2;      // shared experts
constexpr int Bb     = 2;
constexpr int Ss     = 1024;
constexpr int T      = Bb * Ss;      // 2048 tokens
constexpr int NE     = E + NSH;      // 10 "experts" total (8 routed + 2 shared)
constexpr int MAXP   = T;            // max pairs per expert (padded lists)

// ---------------- WMMA types ----------------
typedef __attribute__((ext_vector_type(16))) __bf16 v16bf;
typedef __attribute__((ext_vector_type(2)))  __bf16 v2bf;
typedef __attribute__((ext_vector_type(8)))  float  v8f;

// fp32 -> bf16 (hardware RNE convert; packed fast-path when available)
__device__ __forceinline__ v2bf cvt_pk_bf16(float a, float b) {
#if __has_builtin(__builtin_amdgcn_cvt_pk_bf16_f32)
  return __builtin_amdgcn_cvt_pk_bf16_f32(a, b);
#else
  v2bf r; r[0] = (__bf16)a; r[1] = (__bf16)b; return r;
#endif
}
__device__ __forceinline__ unsigned short bf_bits(float f) {
  union { __bf16 h; unsigned short s; } v;
  v.h = (__bf16)f;
  return v.s;
}

union U128 {            // 16-byte chunk = 8 bf16
  uint4 u;
  __bf16 h[8];
};

// Build a 32x16 B fragment: lane holds 16 consecutive K values of one column
// (column = lane%16, K base = (lane/16)*16), converted fp32 -> bf16.
__device__ __forceinline__ v16bf load_b_frag(const float* p) {
  float4 f0 = ((const float4*)p)[0];
  float4 f1 = ((const float4*)p)[1];
  float4 f2 = ((const float4*)p)[2];
  float4 f3 = ((const float4*)p)[3];
  union { v2bf p2[8]; v16bf v; } b;
  b.p2[0] = cvt_pk_bf16(f0.x, f0.y);
  b.p2[1] = cvt_pk_bf16(f0.z, f0.w);
  b.p2[2] = cvt_pk_bf16(f1.x, f1.y);
  b.p2[3] = cvt_pk_bf16(f1.z, f1.w);
  b.p2[4] = cvt_pk_bf16(f2.x, f2.y);
  b.p2[5] = cvt_pk_bf16(f2.z, f2.w);
  b.p2[6] = cvt_pk_bf16(f3.x, f3.y);
  b.p2[7] = cvt_pk_bf16(f3.z, f3.w);
  return b.v;
}

// Build a 16x32 A fragment from bf16 row-major storage:
// lane row = lane%16, K = kb..kb+15, kb = (lane/16)*16.
__device__ __forceinline__ v16bf load_a_frag(const unsigned short* row, int off) {
  U128 lo, hi;
  lo.u = *(const uint4*)(row + off);
  hi.u = *(const uint4*)(row + off + 8);
  v16bf a;
#pragma unroll
  for (int j = 0; j < 8; ++j) { a[j] = lo.h[j]; a[j + 8] = hi.h[j]; }
  return a;
}

__device__ __forceinline__ float fast_sigmoid(float z) {
  return __builtin_amdgcn_rcpf(1.0f + __expf(-z));
}

// ---------------- kernel 0: init ----------------
__global__ void k_init(const float* __restrict__ x, float* __restrict__ out,
                       unsigned short* __restrict__ xbf,
                       int* __restrict__ ptok, float* __restrict__ pw,
                       int* __restrict__ cnt) {
  long long i = (long long)blockIdx.x * blockDim.x + threadIdx.x;
  if (i < (long long)T * H) {
    out[i] = 0.0f;
    xbf[i] = bf_bits(x[i]);
  }
  if (i < (long long)NE * MAXP) {
    int e = (int)(i / MAXP);
    int t = (int)(i % MAXP);
    if (e >= E) { ptok[i] = t; pw[i] = 1.0f; }   // shared experts: every token, w=1
    else        { ptok[i] = 0; pw[i] = 0.0f; }   // routed: pad (router overwrites)
  }
  if (i < NE) cnt[i] = (i < E) ? 0 : MAXP;
}

// ---------------- kernel 1: router ----------------
__global__ void k_router(const float* __restrict__ x, const float* __restrict__ wr,
                         int* __restrict__ ptok, float* __restrict__ pw,
                         int* __restrict__ cnt) {
  const int wave = threadIdx.x >> 5;
  const int lane = threadIdx.x & 31;
  const int t = blockIdx.x * 8 + wave;
  if (t >= T) return;

  float acc[E];
#pragma unroll
  for (int e = 0; e < E; ++e) acc[e] = 0.0f;

  const float* xr = x + (long long)t * H;
  for (int h = lane; h < H; h += 32) {
    float xv = xr[h];
#pragma unroll
    for (int e = 0; e < E; ++e) acc[e] += xv * wr[e * H + h];
  }
#pragma unroll
  for (int e = 0; e < E; ++e) {
#pragma unroll
    for (int off = 16; off > 0; off >>= 1)
      acc[e] += __shfl_xor(acc[e], off, 32);
  }
  if (lane == 0) {
    float s[E];
#pragma unroll
    for (int e = 0; e < E; ++e) s[e] = fast_sigmoid(acc[e]);
    int e1 = 0;
#pragma unroll
    for (int e = 1; e < E; ++e) if (s[e] > s[e1]) e1 = e;
    int e2 = -1;
#pragma unroll
    for (int e = 0; e < E; ++e)
      if (e != e1 && (e2 < 0 || s[e] > s[e2])) e2 = e;
    float inv = __builtin_amdgcn_rcpf(s[e1] + s[e2]);
    int p1 = atomicAdd(&cnt[e1], 1);
    ptok[e1 * MAXP + p1] = t; pw[e1 * MAXP + p1] = s[e1] * inv;
    int p2 = atomicAdd(&cnt[e2], 1);
    ptok[e2 * MAXP + p2] = t; pw[e2 * MAXP + p2] = s[e2] * inv;
  }
}

// ---------------- kernel 2: gate/up grouped GEMM (WMMA bf16) ----------------
// Block = 256 thr = 8 waves arranged 2(M) x 4(N). Block tile: M=64 rows, N=128 cols.
// Wave tile: 32x32 per matrix (2x2 WMMA accumulators x 2 matrices).
__global__ __launch_bounds__(256) void k_gemm1(
    const unsigned short* __restrict__ xbf,
    const float* __restrict__ wg, const float* __restrict__ wu,
    const float* __restrict__ sg, const float* __restrict__ su,
    unsigned short* __restrict__ act,
    const int* __restrict__ ptok, const int* __restrict__ cnt) {
  const int e     = blockIdx.x >> 5;       // 32 m-tiles per expert
  const int mtile = blockIdx.x & 31;
  const int m0    = mtile * 64;
  if (m0 >= cnt[e]) return;
  const int i0 = blockIdx.y * 128;

  const int wave  = threadIdx.x >> 5;
  const int lane  = threadIdx.x & 31;
  const int wm    = wave & 1;              // 0..1
  const int wn    = wave >> 1;             // 0..3
  const int lhalf = lane >> 4;             // 0..1
  const int lmod  = lane & 15;             // 0..15
  const int kb    = lhalf * 16;            // A K-chunk base
  const int kgb   = lhalf * 16;            // B K-chunk base

  const float* Wg = (e < E) ? wg + (long long)e * I_DIM * H
                            : sg + (long long)(e - E) * I_DIM * H;
  const float* Wu = (e < E) ? wu + (long long)e * I_DIM * H
                            : su + (long long)(e - E) * I_DIM * H;

  const unsigned short* aptr[2];
#pragma unroll
  for (int ms = 0; ms < 2; ++ms) {
    int rl  = m0 + wm * 32 + ms * 16 + lmod;          // A-layout row = lane%16
    int tok = ptok[e * MAXP + rl];
    aptr[ms] = xbf + (long long)tok * H;
  }
  const float* bgp[2]; const float* bup[2];
#pragma unroll
  for (int ns = 0; ns < 2; ++ns) {
    int r = i0 + wn * 32 + ns * 16 + lmod;            // B-layout col = lane%16
    bgp[ns] = Wg + (long long)r * H;
    bup[ns] = Wu + (long long)r * H;
  }

  v8f cg[2][2] = {};
  v8f cu[2][2] = {};

  for (int h0 = 0; h0 < H; h0 += 32) {
    v16bf a[2];
#pragma unroll
    for (int ms = 0; ms < 2; ++ms) a[ms] = load_a_frag(aptr[ms], h0 + kb);

    v16bf bg[2], bu[2];
#pragma unroll
    for (int ns = 0; ns < 2; ++ns) {
      // prefetch the weight stream a few K-steps ahead (global_prefetch_b8)
      __builtin_prefetch(bgp[ns] + h0 + kgb + 256, 0, 1);
      __builtin_prefetch(bup[ns] + h0 + kgb + 256, 0, 1);
      bg[ns] = load_b_frag(bgp[ns] + h0 + kgb);
      bu[ns] = load_b_frag(bup[ns] + h0 + kgb);
    }
#pragma unroll
    for (int ms = 0; ms < 2; ++ms)
#pragma unroll
      for (int ns = 0; ns < 2; ++ns) {
        cg[ms][ns] = __builtin_amdgcn_wmma_f32_16x16x32_bf16(
            false, a[ms], false, bg[ns], (short)0, cg[ms][ns], false, false);
        cu[ms][ns] = __builtin_amdgcn_wmma_f32_16x16x32_bf16(
            false, a[ms], false, bu[ns], (short)0, cu[ms][ns], false, false);
      }
  }

  // epilogue: act = silu(g) * u   (C layout: M = v + 8*lhalf, N = lmod)
#pragma unroll
  for (int ms = 0; ms < 2; ++ms)
#pragma unroll
    for (int ns = 0; ns < 2; ++ns) {
      int col = i0 + wn * 32 + ns * 16 + lmod;
#pragma unroll
      for (int v = 0; v < 8; ++v) {
        int rl = m0 + wm * 32 + ms * 16 + v + 8 * lhalf;
        float g = cg[ms][ns][v];
        float u = cu[ms][ns][v];
        float val = g * fast_sigmoid(g) * u;            // silu(g)*u
        act[(long long)(e * MAXP + rl) * I_DIM + col] = bf_bits(val);
      }
    }
}

// ---------------- kernel 3: down-proj grouped GEMM + weighted scatter ----------------
__global__ __launch_bounds__(256) void k_gemm2(
    const unsigned short* __restrict__ act,
    const float* __restrict__ wd, const float* __restrict__ sd,
    const int* __restrict__ ptok, const float* __restrict__ pw,
    const int* __restrict__ cnt, float* __restrict__ out) {
  const int e     = blockIdx.x >> 5;
  const int mtile = blockIdx.x & 31;
  const int m0    = mtile * 64;
  if (m0 >= cnt[e]) return;
  const int h0out = blockIdx.y * 128;

  const int wave  = threadIdx.x >> 5;
  const int lane  = threadIdx.x & 31;
  const int wm    = wave & 1;
  const int wn    = wave >> 1;
  const int lhalf = lane >> 4;
  const int lmod  = lane & 15;
  const int kb    = lhalf * 16;
  const int kgb   = lhalf * 16;

  const float* Wd = (e < E) ? wd + (long long)e * H * I_DIM
                            : sd + (long long)(e - E) * H * I_DIM;

  const unsigned short* aptr[2];
#pragma unroll
  for (int ms = 0; ms < 2; ++ms) {
    int rl = m0 + wm * 32 + ms * 16 + lmod;
    aptr[ms] = act + (long long)(e * MAXP + rl) * I_DIM;
  }
  const float* bp[2];
#pragma unroll
  for (int ns = 0; ns < 2; ++ns) {
    int r = h0out + wn * 32 + ns * 16 + lmod;
    bp[ns] = Wd + (long long)r * I_DIM;
  }

  v8f c[2][2] = {};

  for (int k0 = 0; k0 < I_DIM; k0 += 32) {
    v16bf a[2];
#pragma unroll
    for (int ms = 0; ms < 2; ++ms) a[ms] = load_a_frag(aptr[ms], k0 + kb);

    v16bf b[2];
#pragma unroll
    for (int ns = 0; ns < 2; ++ns) {
      __builtin_prefetch(bp[ns] + k0 + kgb + 256, 0, 1);
      b[ns] = load_b_frag(bp[ns] + k0 + kgb);
    }
#pragma unroll
    for (int ms = 0; ms < 2; ++ms)
#pragma unroll
      for (int ns = 0; ns < 2; ++ns)
        c[ms][ns] = __builtin_amdgcn_wmma_f32_16x16x32_bf16(
            false, a[ms], false, b[ns], (short)0, c[ms][ns], false, false);
  }

  // epilogue: out[token, h] += w(pair) * c
#pragma unroll
  for (int ms = 0; ms < 2; ++ms)
#pragma unroll
    for (int ns = 0; ns < 2; ++ns) {
      int col = h0out + wn * 32 + ns * 16 + lmod;
#pragma unroll
      for (int v = 0; v < 8; ++v) {
        int rl   = m0 + wm * 32 + ms * 16 + v + 8 * lhalf;
        int pidx = e * MAXP + rl;
        int tok  = ptok[pidx];
        float w  = pw[pidx];                       // pad rows have w==0
        atomicAdd(&out[(long long)tok * H + col], w * c[ms][ns][v]);
      }
    }
}

// ---------------- host launch ----------------
extern "C" void kernel_launch(void* const* d_in, const int* in_sizes, int n_in,
                              void* d_out, int out_size, void* d_ws, size_t ws_size,
                              hipStream_t stream) {
  const float* x  = (const float*)d_in[0];
  const float* wr = (const float*)d_in[1];
  const float* wg = (const float*)d_in[2];
  const float* wu = (const float*)d_in[3];
  const float* wd = (const float*)d_in[4];
  const float* sg = (const float*)d_in[5];
  const float* su = (const float*)d_in[6];
  const float* sd = (const float*)d_in[7];
  float* out = (float*)d_out;

  // workspace layout
  unsigned short* xbf = (unsigned short*)d_ws;                       // T*H bf16
  unsigned short* act = xbf + (long long)T * H;                      // NE*MAXP*I bf16
  int*   ptok = (int*)(act + (long long)NE * MAXP * I_DIM);          // NE*MAXP int
  float* pwv  = (float*)(ptok + NE * MAXP);                          // NE*MAXP float
  int*   cnt  = (int*)(pwv + NE * MAXP);                             // NE int

  const int initN = T * H;                                           // covers all init work
  k_init<<<(initN + 255) / 256, 256, 0, stream>>>(x, out, xbf, ptok, pwv, cnt);
  k_router<<<T / 8, 256, 0, stream>>>(x, wr, ptok, pwv, cnt);
  k_gemm1<<<dim3(NE * 32, I_DIM / 128), 256, 0, stream>>>(xbf, wg, wu, sg, su,
                                                          act, ptok, cnt);
  k_gemm2<<<dim3(NE * 32, H / 128), 256, 0, stream>>>(act, wd, sd, ptok, pwv,
                                                      cnt, out);
}
